// EmbeddingQuantizer_56908316672010
// MI455X (gfx1250) — matrix operand order repaired and checked by reading, not compiled, not run on hardware
//
#include <hip/hip_runtime.h>
#include <hip/hip_bf16.h>

typedef __attribute__((ext_vector_type(16))) __bf16 v16bf;
typedef __attribute__((ext_vector_type(8)))  __bf16 v8bf;
typedef __attribute__((ext_vector_type(8)))  float  v8f;
typedef __attribute__((ext_vector_type(4)))  unsigned int v4u;
typedef __attribute__((ext_vector_type(8)))  int v8i;
typedef __attribute__((ext_vector_type(4)))  int v4i;

#define BSZ    16
#define CDIM   256
#define HDIM   32
#define WDIM   32
#define HW     (HDIM * WDIM)           // 1024
#define NTOT   (BSZ * HW)              // 16384 rows
#define KCODES 8192
#define DDIM   256
#define NTILES (KCODES / 16)           // 512 k-tiles total
#define KSPLIT 8                       // k-range split across blocks
#define SEGTILES (NTILES / KSPLIT)     // 64 tiles per block
#define WAVES  8                       // waves per block in vq_argmin
#define ROWSPW 32                      // rows per wave (2 strips of 16)
// LDS row pitch for one code row: 256 bf16 (=128 dwords) + 4 pad dwords (TDM pad)
#define BPITCH 528                     // bytes: 132 dwords

// ---------------------------------------------------------------------------
// TDM issue: load one 16x256 bf16 tile (16 codebook rows) into LDS with a
// 4-dword pad after every 128 dwords (one row) -> LDS pitch 528 B, which makes
// the per-column ds_load_b128 pattern bank-conflict free (lane l -> bank 4l).
// D# layout per CDNA5 ISA ch.8 (group0: flags/lds/global addr; group1: dims).
// ---------------------------------------------------------------------------
__device__ __forceinline__ void tdm_issue_tile(unsigned lds_byte_off,
                                               const __bf16* gsrc) {
    unsigned long long ga = (unsigned long long)gsrc;
    v4u g0 = { 1u,                                   // count=1 (valid user D#)
               lds_byte_off,                         // lds_addr (bytes)
               (unsigned)(ga & 0xffffffffu),         // global_addr[31:0]
               (unsigned)((ga >> 32) & 0x1ffffffu)   // global_addr[56:32]
                   | (2u << 30) };                   // type=2 ("image")
    v8i g1 = { (int)((1u << 16)        //  data_size = 1 -> 2-byte elements
                   | (1u << 20)        //  pad_enable
                   | (6u << 22)        //  pad_interval = 6 -> every 128 dwords
                   | (3u << 25)),      //  pad_amount   = 3 -> 4 dwords
               (int)(256u << 16),      //  tensor_dim0 = 256 (bits 63:48)
               (int)(8192u << 16),     //  tensor_dim1 = 8192 (bits 95:80)
               (int)(256u << 16),      //  tile_dim0 = 256 (bits 127:112)
               16,                     //  tile_dim1 = 16, tile_dim2 = 0
               256,                    //  tensor_dim0_stride = 256
               0, 0 };                 //  tensor_dim1_stride = 0 (unused, 2D)
    v4i gz = { 0, 0, 0, 0 };
#if defined(__clang_major__) && (__clang_major__ >= 23)
    v8i gz8 = { 0, 0, 0, 0, 0, 0, 0, 0 };
    __builtin_amdgcn_tensor_load_to_lds(g0, g1, gz, gz, gz8, 0);
#else
    __builtin_amdgcn_tensor_load_to_lds(g0, g1, gz, gz, 0);
#endif
}

__device__ __forceinline__ unsigned lds_offset_of(const void* p) {
    return (unsigned)(unsigned long long)p;
}

// float score -> order-preserving u32 (works for negatives), packed with index
// so that u64 min == (min score, then min index): exact argmin tie-break.
__device__ __forceinline__ unsigned long long pack_score(float s, int i) {
    unsigned u = __float_as_uint(s);
    u = (u & 0x80000000u) ? ~u : (u | 0x80000000u);
    return ((unsigned long long)u << 32) | (unsigned)i;
}

// ---------------------------------------------------------------------------
// Kernel 0: init per-row winner records and loss accumulator.
// ---------------------------------------------------------------------------
__global__ void vq_init(unsigned long long* __restrict__ best,
                        float* __restrict__ loss) {
    const int i = blockIdx.x * 256 + threadIdx.x;
    if (i < NTOT) best[i] = ~0ull;
    if (i == 0) *loss = 0.0f;
}

// ---------------------------------------------------------------------------
// Kernel 1: transpose+cast x [B,C,H,W] (f32) -> Xb [N, D] (bf16), N = b*HW+hw
// ---------------------------------------------------------------------------
__global__ void vq_transpose_x(const float* __restrict__ x,
                               __bf16* __restrict__ Xb) {
    __shared__ float tile[32][33];
    const int b   = blockIdx.z;
    const int c0  = blockIdx.y * 32;
    const int hw0 = blockIdx.x * 32;
    const int tx  = threadIdx.x;   // 0..31
    const int ty  = threadIdx.y;   // 0..7
    const float* xb = x + (size_t)b * CDIM * HW;
#pragma unroll
    for (int i = 0; i < 32; i += 8) {
        tile[ty + i][tx] = xb[(size_t)(c0 + ty + i) * HW + hw0 + tx];
    }
    __syncthreads();
    __bf16* out = Xb + (size_t)b * HW * DDIM;
#pragma unroll
    for (int i = 0; i < 32; i += 8) {
        const int hw = hw0 + ty + i;
        out[(size_t)hw * DDIM + c0 + tx] = (__bf16)tile[tx][ty + i];
    }
}

// ---------------------------------------------------------------------------
// Kernel 2: codebook f32 -> bf16 copy + per-row squared norms (f32).
// ---------------------------------------------------------------------------
__global__ void vq_prep_codebook(const float* __restrict__ cb,
                                 __bf16* __restrict__ Cb,
                                 float* __restrict__ ys) {
    const int k = blockIdx.x;
    const int t = threadIdx.x;     // 0..63
    const float* row = cb + (size_t)k * DDIM;
    __bf16* orow = Cb + (size_t)k * DDIM;
    float s = 0.0f;
#pragma unroll
    for (int i = 0; i < 4; ++i) {
        const float v = row[t + 64 * i];
        s += v * v;
        orow[t + 64 * i] = (__bf16)v;
    }
    __shared__ float red[64];
    red[t] = s;
    __syncthreads();
#pragma unroll
    for (int off = 32; off > 0; off >>= 1) {
        if (t < off) red[t] += red[t + off];
        __syncthreads();
    }
    if (t == 0) ys[k] = red[0];
}

// ---------------------------------------------------------------------------
// Kernel 3: GEMM + argmin core.
// Grid = (row-blocks of 256 rows) x (KSPLIT k-segments).  8 waves per block;
// each wave owns TWO 16-row A strips (128 VGPRs resident) and runs two
// independent WMMA accumulator chains off each shared B fragment -> 16 WMMAs
// per 8 KB LDS tile read (2x arithmetic intensity vs LDS).  The B tile is
// DMA'd into LDS by the TDM, double-buffered and TENSORcnt-paced.  Per-block
// winners merge globally via packed atomic_min_u64.
// ---------------------------------------------------------------------------
__global__ void __launch_bounds__(32 * WAVES)
vq_argmin(const __bf16* __restrict__ Xb,
          const __bf16* __restrict__ Cb,
          const float* __restrict__ ys,
          unsigned long long* __restrict__ best) {
    __shared__ __align__(16) char Bt[2][16 * BPITCH];       // double-buffered B tile
    __shared__ unsigned long long sbest[WAVES][ROWSPW][16]; // packed reduce buf

    const int lane = threadIdx.x & 31;
    const int wave = threadIdx.x >> 5;
    const int n0   = blockIdx.x * (WAVES * ROWSPW) + wave * ROWSPW;
    const int seg  = blockIdx.y;                 // k-segment
    const int col  = lane & 15;
    const int hi   = lane >> 4;

    // ---- preload A fragments for both strips ------------------------------
    const __bf16* arowA = Xb + (size_t)(n0 + col) * DDIM;
    const __bf16* arowB = Xb + (size_t)(n0 + 16 + col) * DDIM;
    v16bf afragA[8], afragB[8];
#pragma unroll
    for (int dc = 0; dc < 8; ++dc) {
        const int d0 = dc * 32;
        const v8bf a0 = *(const v8bf*)(arowA + d0 + hi * 16);
        const v8bf a1 = *(const v8bf*)(arowA + d0 + hi * 16 + 8);
        const v8bf b0 = *(const v8bf*)(arowB + d0 + hi * 16);
        const v8bf b1 = *(const v8bf*)(arowB + d0 + hi * 16 + 8);
        v16bf a, b;
#pragma unroll
        for (int i = 0; i < 8; ++i) {
            a[i] = a0[i]; a[8 + i] = a1[i];
            b[i] = b0[i]; b[8 + i] = b1[i];
        }
        afragA[dc] = a;
        afragB[dc] = b;
    }

    float bestdA[8], bestdB[8];
    int   bestiA[8], bestiB[8];
#pragma unroll
    for (int v = 0; v < 8; ++v) {
        bestdA[v] = 3.0e38f; bestiA[v] = 0;
        bestdB[v] = 3.0e38f; bestiB[v] = 0;
    }

    const __bf16* segbase = Cb + (size_t)seg * SEGTILES * 16 * DDIM;

    // ---- prologue: DMA tile 0 of this segment into buffer 0 ---------------
    if (wave == 0) {
        tdm_issue_tile(lds_offset_of(&Bt[0][0]), segbase);
    }

    for (int t = 0; t < SEGTILES; ++t) {
        const int buf = t & 1;
        if (wave == 0) {
            if (t + 1 < SEGTILES) {
                tdm_issue_tile(lds_offset_of(&Bt[buf ^ 1][0]),
                               segbase + (size_t)(t + 1) * 16 * DDIM);
                __builtin_amdgcn_s_wait_tensorcnt(1);   // tile t complete
            } else {
                __builtin_amdgcn_s_wait_tensorcnt(0);
            }
        }
        __syncthreads();   // publish tile t to all waves

        const char* bbase = &Bt[buf][0] + col * BPITCH + hi * 32;
        v8f accA = {}, accB = {};
#pragma unroll
        for (int dc = 0; dc < 8; ++dc) {
            const v16bf bfrag = *(const v16bf*)(bbase + dc * 64);
            accA = __builtin_amdgcn_wmma_f32_16x16x32_bf16(
                false, afragA[dc], false, bfrag, (short)0, accA, false, false);
            accB = __builtin_amdgcn_wmma_f32_16x16x32_bf16(
                false, afragB[dc], false, bfrag, (short)0, accB, false, false);
        }

        const int k0 = (seg * SEGTILES + t) * 16;
        const float yk = ys[k0 + col];   // ||e||^2 ; ||x||^2 is row-constant
#pragma unroll
        for (int v = 0; v < 8; ++v) {
            const float sA = yk - 2.0f * accA[v];
            const float sB = yk - 2.0f * accB[v];
            if (sA < bestdA[v]) { bestdA[v] = sA; bestiA[v] = k0 + col; }
            if (sB < bestdB[v]) { bestdB[v] = sB; bestiB[v] = k0 + col; }
        }
        __syncthreads();   // all waves done with buf before TDM overwrites it
    }

    // ---- reduce: pack (score,idx); min over the 16 column-lanes -----------
#pragma unroll
    for (int v = 0; v < 8; ++v) {
        sbest[wave][v + 8 * hi][col]      = pack_score(bestdA[v], bestiA[v]);
        sbest[wave][16 + v + 8 * hi][col] = pack_score(bestdB[v], bestiB[v]);
    }
    __syncthreads();
    {
        // lane r (0..31) reduces row r of this wave's 32-row strip
        unsigned long long m = sbest[wave][lane][0];
#pragma unroll
        for (int j = 1; j < 16; ++j) {
            const unsigned long long c = sbest[wave][lane][j];
            m = c < m ? c : m;
        }
        atomicMin(&best[n0 + lane], m);   // merge across KSPLIT segments
    }
}

// ---------------------------------------------------------------------------
// Kernel 4: gather quantized = codebook[idx] back into [B,C,H,W] (f32),
// emit idx as f32, and accumulate mean((x - q)^2) via atomicAdd.
// ---------------------------------------------------------------------------
__global__ void vq_gather_loss(const float* __restrict__ x,
                               const float* __restrict__ cb,
                               const unsigned long long* __restrict__ best,
                               float* __restrict__ q,
                               float* __restrict__ idx_f,
                               float* __restrict__ loss) {
    const int n  = blockIdx.x;
    const int t  = threadIdx.x;    // 0..63
    const int b  = n / HW;
    const int hw = n % HW;
    const int idx = (int)(unsigned)(best[n] & 0xffffffffu);
    if (t == 0) idx_f[n] = (float)idx;
    const float* crow = cb + (size_t)idx * DDIM;
    const float* xp = x + (size_t)b * CDIM * HW + hw;
    float*       qp = q + (size_t)b * CDIM * HW + hw;
    float s = 0.0f;
#pragma unroll
    for (int i = 0; i < 4; ++i) {
        const int c = t + 64 * i;
        const float e  = crow[c];
        const float xv = xp[(size_t)c * HW];
        qp[(size_t)c * HW] = e;
        const float d = xv - e;
        s += d * d;
    }
    __shared__ float red[64];
    red[t] = s;
    __syncthreads();
#pragma unroll
    for (int off = 32; off > 0; off >>= 1) {
        if (t < off) red[t] += red[t + off];
        __syncthreads();
    }
    if (t == 0)
        atomicAdd(loss, red[0] * (1.0f / (float)((size_t)BSZ * CDIM * HW)));
}

// ---------------------------------------------------------------------------
extern "C" void kernel_launch(void* const* d_in, const int* in_sizes, int n_in,
                              void* d_out, int out_size, void* d_ws, size_t ws_size,
                              hipStream_t stream) {
    const float* x  = (const float*)d_in[0];   // [16,256,32,32] f32
    const float* cb = (const float*)d_in[1];   // [8192,256]     f32

    // outputs concatenated: quantized [B,C,H,W] f32 | idx [N] (as f32) | loss
    float* q     = (float*)d_out;
    float* idx_f = q + (size_t)BSZ * CDIM * HW;
    float* loss  = idx_f + NTOT;

    // workspace: Xb bf16 (8MB) | Cb bf16 (4MB) | ys f32 (32KB) | best u64 (128KB)
    char* ws = (char*)d_ws;
    __bf16* Xb = (__bf16*)ws;  ws += (size_t)NTOT   * DDIM * sizeof(__bf16);
    __bf16* Cb = (__bf16*)ws;  ws += (size_t)KCODES * DDIM * sizeof(__bf16);
    float*  ys = (float*)ws;   ws += (size_t)KCODES * sizeof(float);
    unsigned long long* best = (unsigned long long*)ws;

    vq_init<<<(NTOT + 255) / 256, 256, 0, stream>>>(best, loss);

    dim3 tb(32, 8, 1);
    dim3 tg(HW / 32, CDIM / 32, BSZ);
    vq_transpose_x<<<tg, tb, 0, stream>>>(x, Xb);
    vq_prep_codebook<<<KCODES, 64, 0, stream>>>(cb, Cb, ys);

    dim3 gg(NTOT / (WAVES * ROWSPW), KSPLIT, 1);   // 64 x 8 = 512 blocks
    vq_argmin<<<gg, 32 * WAVES, 0, stream>>>(Xb, Cb, ys, best);

    vq_gather_loss<<<NTOT, 64, 0, stream>>>(x, cb, best, q, idx_f, loss);
}
